// BertSelfAttention_31825707663511
// MI455X (gfx1250) — compile-verified, matrix operand order checked by reading, not run on hardware
//
#include <hip/hip_runtime.h>
#include <hip/hip_bf16.h>
#include <cstddef>

typedef _Float16 v16h __attribute__((ext_vector_type(16)));
typedef _Float16 v8h  __attribute__((ext_vector_type(8)));
typedef float    v8f  __attribute__((ext_vector_type(8)));

#define S_LEN 2048
#define BATCH 4
#define HID   1024
#define NHEAD 16
#define DHEAD 64
#define MROWS (S_LEN * BATCH)   // 8192

// ---------------------------------------------------------------------------
// fp32 -> f16 conversion (grid-stride)
// ---------------------------------------------------------------------------
__global__ void cvt_f32_f16(const float* __restrict__ in,
                            _Float16* __restrict__ out, int n) {
  int i = blockIdx.x * blockDim.x + threadIdx.x;
  int stride = gridDim.x * blockDim.x;
  for (; i < n; i += stride) out[i] = (_Float16)in[i];
}

// ---------------------------------------------------------------------------
// NT GEMM, register-blocked 32(M) x 64(N) per wave:
//   out[m,n] = sum_k A[m,k] * W[n,k] + bias[n]
// A: [8192 x 1024] f16 row-major (m = s*B + b), W: [1024 x 1024] f16 row-major.
// Per 32-wide K step: 2 A frags + 4 B frags (12 b128) -> 8 WMMAs.
// TRANSPOSED==0: plain [m][HID] f16 output (used for Q, K).
// TRANSPOSED==1: [bh][d][s] f16 output (V^T, keys contiguous for PV WMMA).
// All epilogue stores are base + compile-time-immediate (branch-free).
// ---------------------------------------------------------------------------
template <int TRANSPOSED>
__global__ __launch_bounds__(256) void gemm_qkv(
    const _Float16* __restrict__ A,
    const _Float16* __restrict__ W,
    const float* __restrict__ bias,
    _Float16* __restrict__ out) {
  const int lane  = threadIdx.x & 31;
  const int wave  = threadIdx.x >> 5;
  const int tile  = blockIdx.x * 8 + wave;   // 256 mTiles x 16 nTiles = 4096
  const int mT    = tile >> 4;
  const int nT    = tile & 15;
  const int m0    = mT * 32;
  const int n0    = nT * 64;
  const int l15   = lane & 15;
  const int kb    = (lane >= 16) ? 8 : 0;    // K sub-offset per WMMA A/B layout
  const int half8 = kb;                      // row offset for C layout

  const _Float16* ap0 = A + (size_t)(m0 + l15) * HID + kb;
  const _Float16* ap1 = ap0 + (size_t)16 * HID;
  const _Float16* bp0 = W + (size_t)(n0 + l15) * HID + kb;

  v8f acc[2][4] = {};
  for (int k = 0; k < HID; k += 32) {
    v16h af[2], bf[4];
    ((v8h*)&af[0])[0] = *(const v8h*)(ap0 + k);
    ((v8h*)&af[0])[1] = *(const v8h*)(ap0 + k + 16);
    ((v8h*)&af[1])[0] = *(const v8h*)(ap1 + k);
    ((v8h*)&af[1])[1] = *(const v8h*)(ap1 + k + 16);
#pragma unroll
    for (int ni = 0; ni < 4; ++ni) {
      const _Float16* bp = bp0 + (size_t)(16 * ni) * HID + k;
      ((v8h*)&bf[ni])[0] = *(const v8h*)(bp);
      ((v8h*)&bf[ni])[1] = *(const v8h*)(bp + 16);
    }
#pragma unroll
    for (int mi = 0; mi < 2; ++mi)
#pragma unroll
      for (int ni = 0; ni < 4; ++ni)
        acc[mi][ni] = __builtin_amdgcn_wmma_f32_16x16x32_f16(
            false, af[mi], false, bf[ni], (short)0, acc[mi][ni], false, false);
  }

  // C layout: lane -> col n = n0 + 16*ni + l15, vgpr r -> row m = .. + r + half8
  float bn[4];
#pragma unroll
  for (int ni = 0; ni < 4; ++ni) bn[ni] = bias[n0 + 16 * ni + l15];

  if (TRANSPOSED == 0) {
    // out[m][n], per-lane base + constant offsets
    _Float16* ob = out + (size_t)(m0 + half8) * HID + n0 + l15;
#pragma unroll
    for (int mi = 0; mi < 2; ++mi)
#pragma unroll
      for (int ni = 0; ni < 4; ++ni)
#pragma unroll
        for (int r = 0; r < 8; ++r)
          ob[(size_t)(16 * mi + r) * HID + 16 * ni] =
              (_Float16)(acc[mi][ni][r] + bn[ni]);
  } else {
    // V^T: out[((bidx*16 + nT)*64 + d)*2048 + s], with
    //   bidx = r&3 (compile-time), s = m0/4 + 4*mi + (kb>>2) + (r>>2)
    // head == nT is uniform because n0 = nT*64 spans exactly one head.
#pragma unroll
    for (int ni = 0; ni < 4; ++ni) {
      const int d = 16 * ni + l15;
      _Float16* vb = out + (size_t)(nT * DHEAD + d) * S_LEN + (m0 >> 2) + (kb >> 2);
#pragma unroll
      for (int mi = 0; mi < 2; ++mi)
#pragma unroll
        for (int r = 0; r < 8; ++r)
          vb[(size_t)(r & 3) * (NHEAD * DHEAD * S_LEN) + 4 * mi + (r >> 2)] =
              (_Float16)(acc[mi][ni][r] + bn[ni]);
    }
  }
}

// ---------------------------------------------------------------------------
// Flash attention: one wave per (bh, 16-query tile). 64-key chunks:
//   16 WMMAs per chunk (8 QK^T + 8 PV), deferred row-sum reduction.
// Qm,Km: [8192][1024] f16 (row m = s*B + b, col = h*64 + d)
// VT:    [bh][64][2048] f16;  mask: [B][S] f32;  out: [S,B,H] f32
// ---------------------------------------------------------------------------
__global__ __launch_bounds__(256) void flash_attn(
    const _Float16* __restrict__ Qm,
    const _Float16* __restrict__ Km,
    const _Float16* __restrict__ VT,
    const float* __restrict__ mask,
    float* __restrict__ out) {
  __shared__ __align__(16) _Float16 pbuf[8][16][64];   // per-wave P staging, 16KB

  const int lane = threadIdx.x & 31;
  const int wave = threadIdx.x >> 5;
  const int tile = blockIdx.x * 8 + wave;
  const int bh   = tile >> 7;           // S/16 = 128 query tiles
  const int qt   = tile & 127;
  const int b    = bh >> 4;
  const int h    = bh & 15;
  const int q0   = qt * 16;
  const int l15  = lane & 15;
  const int kb   = (lane >= 16) ? 8 : 0;

  // Q A-fragments: row m = (q0+l15)*B + b, cols h*64 + (0..63)
  const _Float16* qrow =
      Qm + (size_t)((q0 + l15) * BATCH + b) * HID + h * DHEAD;
  v16h qa0, qa1;
  ((v8h*)&qa0)[0] = *(const v8h*)(qrow + kb);
  ((v8h*)&qa0)[1] = *(const v8h*)(qrow + kb + 16);
  ((v8h*)&qa1)[0] = *(const v8h*)(qrow + 32 + kb);
  ((v8h*)&qa1)[1] = *(const v8h*)(qrow + 32 + kb + 16);

  float mrow[8], lrow[8];   // lrow holds per-LANE partial sums (reduced at end)
#pragma unroll
  for (int r = 0; r < 8; ++r) { mrow[r] = -1e30f; lrow[r] = 0.0f; }
  v8f o[4] = {};

  const float* maskb = mask + b * S_LEN;
  const _Float16* Kb   = Km + (size_t)b * HID + h * DHEAD;   // + key*B*HID
  const _Float16* VTbh = VT + (size_t)bh * DHEAD * S_LEN;

  for (int kc = 0; kc < S_LEN; kc += 64) {
    // ---- scores: four 16x16 tiles over 64 keys ----
    v8f s[4] = {};
    const _Float16* kbase = Kb + (size_t)(kc + l15) * BATCH * HID;
    __builtin_prefetch(kbase + (size_t)64 * BATCH * HID, 0, 0);  // next K chunk
#pragma unroll
    for (int j = 0; j < 4; ++j) {
      const _Float16* krow = kbase + (size_t)(16 * j) * BATCH * HID;
      v16h kf0, kf1;
      ((v8h*)&kf0)[0] = *(const v8h*)(krow + kb);
      ((v8h*)&kf0)[1] = *(const v8h*)(krow + kb + 16);
      ((v8h*)&kf1)[0] = *(const v8h*)(krow + 32 + kb);
      ((v8h*)&kf1)[1] = *(const v8h*)(krow + 32 + kb + 16);
      s[j] = __builtin_amdgcn_wmma_f32_16x16x32_f16(false, qa0, false, kf0,
                                                    (short)0, s[j], false, false);
      s[j] = __builtin_amdgcn_wmma_f32_16x16x32_f16(false, qa1, false, kf1,
                                                    (short)0, s[j], false, false);
    }

    float mk[4];
#pragma unroll
    for (int j = 0; j < 4; ++j) mk[j] = maskb[kc + 16 * j + l15];

    // ---- online softmax; row max shuffled, row sum kept per-lane ----
#pragma unroll
    for (int r = 0; r < 8; ++r) {
      float a0 = s[0][r] * 0.125f + mk[0];   // 1/sqrt(64)
      float a1 = s[1][r] * 0.125f + mk[1];
      float a2 = s[2][r] * 0.125f + mk[2];
      float a3 = s[3][r] * 0.125f + mk[3];
      float t  = fmaxf(fmaxf(a0, a1), fmaxf(a2, a3));
      t = fmaxf(t, __shfl_xor(t, 1));
      t = fmaxf(t, __shfl_xor(t, 2));
      t = fmaxf(t, __shfl_xor(t, 4));
      t = fmaxf(t, __shfl_xor(t, 8));
      const float newm = fmaxf(mrow[r], t);
      const float corr = __expf(mrow[r] - newm);
      mrow[r] = newm;
      const float p0 = __expf(a0 - newm);
      const float p1 = __expf(a1 - newm);
      const float p2 = __expf(a2 - newm);
      const float p3 = __expf(a3 - newm);
      lrow[r] = lrow[r] * corr + (p0 + p1) + (p2 + p3);
      o[0][r] *= corr; o[1][r] *= corr; o[2][r] *= corr; o[3][r] *= corr;
      const int row = r + kb;               // kb doubles as the +8 row offset
      pbuf[wave][row][l15]      = (_Float16)p0;
      pbuf[wave][row][16 + l15] = (_Float16)p1;
      pbuf[wave][row][32 + l15] = (_Float16)p2;
      pbuf[wave][row][48 + l15] = (_Float16)p3;
    }
    asm volatile("" ::: "memory");  // keep DS loads after DS stores (HW in-order per wave)

    // P A-fragments: keys kc..kc+31 and kc+32..kc+63
    v16h pa0, pa1;
    ((v8h*)&pa0)[0] = *(const v8h*)&pbuf[wave][l15][kb];
    ((v8h*)&pa0)[1] = *(const v8h*)&pbuf[wave][l15][16 + kb];
    ((v8h*)&pa1)[0] = *(const v8h*)&pbuf[wave][l15][32 + kb];
    ((v8h*)&pa1)[1] = *(const v8h*)&pbuf[wave][l15][48 + kb];

    // ---- P (16x64) @ V (64x64): four 16x16 N-tiles over DH ----
#pragma unroll
    for (int t = 0; t < 4; ++t) {
      const _Float16* vrow = VTbh + (size_t)(t * 16 + l15) * S_LEN + kc;
      if (t == 0) __builtin_prefetch(vrow + 64, 0, 0);     // next V chunk
      v16h vf0, vf1;
      ((v8h*)&vf0)[0] = *(const v8h*)(vrow + kb);
      ((v8h*)&vf0)[1] = *(const v8h*)(vrow + kb + 16);
      ((v8h*)&vf1)[0] = *(const v8h*)(vrow + 32 + kb);
      ((v8h*)&vf1)[1] = *(const v8h*)(vrow + 48 + kb);
      o[t] = __builtin_amdgcn_wmma_f32_16x16x32_f16(false, pa0, false, vf0,
                                                    (short)0, o[t], false, false);
      o[t] = __builtin_amdgcn_wmma_f32_16x16x32_f16(false, pa1, false, vf1,
                                                    (short)0, o[t], false, false);
    }
  }

  // ---- epilogue: reduce per-lane sums across the 16-lane half, write ctx ----
#pragma unroll
  for (int r = 0; r < 8; ++r) {
    float rs = lrow[r];
    rs += __shfl_xor(rs, 1);
    rs += __shfl_xor(rs, 2);
    rs += __shfl_xor(rs, 4);
    rs += __shfl_xor(rs, 8);
    const float inv = 1.0f / rs;
    const int srow = q0 + r + kb;
    float* op = out + ((size_t)srow * BATCH + b) * HID + h * DHEAD + l15;
    op[0]  = o[0][r] * inv;
    op[16] = o[1][r] * inv;
    op[32] = o[2][r] * inv;
    op[48] = o[3][r] * inv;
  }
}

// ---------------------------------------------------------------------------
extern "C" void kernel_launch(void* const* d_in, const int* in_sizes, int n_in,
                              void* d_out, int out_size, void* d_ws, size_t ws_size,
                              hipStream_t stream) {
  const float* hs   = (const float*)d_in[0];   // [S,B,H]
  const float* mask = (const float*)d_in[1];   // [B,1,1,S]
  const float* Wq   = (const float*)d_in[2];
  const float* bq   = (const float*)d_in[3];
  const float* Wk   = (const float*)d_in[4];
  const float* bk   = (const float*)d_in[5];
  const float* Wv   = (const float*)d_in[6];
  const float* bv   = (const float*)d_in[7];
  float* out = (float*)d_out;

  const size_t HS_N  = (size_t)MROWS * HID;                      // 8388608
  const size_t W_N   = (size_t)HID * HID;                        // 1048576
  const size_t QKV_N = (size_t)BATCH * NHEAD * S_LEN * DHEAD;    // 8388608

  _Float16* hs16 = (_Float16*)d_ws;
  _Float16* wq16 = hs16 + HS_N;
  _Float16* wk16 = wq16 + W_N;
  _Float16* wv16 = wk16 + W_N;
  _Float16* Qh   = wv16 + W_N;   // [8192][1024]
  _Float16* Kh   = Qh + QKV_N;   // [8192][1024]
  _Float16* VTh  = Kh + QKV_N;   // [64][64][2048]

  cvt_f32_f16<<<4096, 256, 0, stream>>>(hs, hs16, (int)HS_N);
  cvt_f32_f16<<<1024, 256, 0, stream>>>(Wq, wq16, (int)W_N);
  cvt_f32_f16<<<1024, 256, 0, stream>>>(Wk, wk16, (int)W_N);
  cvt_f32_f16<<<1024, 256, 0, stream>>>(Wv, wv16, (int)W_N);

  // 256 x 16 = 4096 32x64 tiles / 8 waves per block
  gemm_qkv<0><<<512, 256, 0, stream>>>(hs16, wq16, bq, Qh);
  gemm_qkv<0><<<512, 256, 0, stream>>>(hs16, wk16, bk, Kh);
  gemm_qkv<1><<<512, 256, 0, stream>>>(hs16, wv16, bv, VTh);

  // 64 (b,h) * 128 query tiles = 8192 waves / 8 per block
  flash_attn<<<1024, 256, 0, stream>>>(Qh, Kh, VTh, mask, out);
}
